// ContextAwareGATv2_GRU_V4_16698832847409
// MI455X (gfx1250) — compile-verified, matrix operand order checked by reading, not run on hardware
//
#include <hip/hip_runtime.h>
#include <hip/hip_bf16.h>
#include <math.h>

#define N_NODES 50000
#define N_EDGES 800000
#define EE      (N_EDGES + N_NODES)   // with self loops
#define BATCH   4096
#define TMAX    50
#define FSEQ    16

typedef __attribute__((ext_vector_type(16))) __bf16 bf16x16;
typedef __attribute__((ext_vector_type(8)))  float  f32x8;

// ---------------------------------------------------------------------------
// WMMA fragment helpers (CDNA5 wave32, 16x16x32 bf16 -> f32)
// A (16x32, MxK): lane = row (lane&15), element i covers
//   k = ((i<8)? i : i+8) + 8*(lane>>4)           [ISA 7.12.2 16-bit A layout]
// B (32x16, KxN): mirrored -- lane = column, same k mapping.
// C/D (16x16 f32): n = lane&15, m = v + 8*(lane>>4).
// ---------------------------------------------------------------------------
static __device__ __forceinline__ int kmap(int i, int lane) {
  return ((i < 8) ? i : (i + 8)) + ((lane >> 4) << 3);
}

static __device__ __forceinline__ bf16x16
load_a_f32(const float* __restrict__ X, int m0, int k0, int ld, int lane) {
  const float* row = X + (size_t)(m0 + (lane & 15)) * ld + k0;
  bf16x16 a;
#pragma unroll
  for (int i = 0; i < 16; ++i) a[i] = (__bf16)row[kmap(i, lane)];
  return a;
}

// W is [Nout, K] row-major (PyTorch layout); B[k][n] = W[n][k]. No range check.
static __device__ __forceinline__ bf16x16
load_b_w(const float* __restrict__ W, int n0, int k0, int ld, int lane) {
  const float* row = W + (size_t)(n0 + (lane & 15)) * ld + k0;
  bf16x16 b;
#pragma unroll
  for (int i = 0; i < 16; ++i) b[i] = (__bf16)row[kmap(i, lane)];
  return b;
}

// Guarded variant (partial column tiles only; used by the N=1 head).
static __device__ __forceinline__ bf16x16
load_b_w_guard(const float* __restrict__ W, int n0, int k0, int ld, int nmax, int lane) {
  int n = n0 + (lane & 15);
  bf16x16 b;
  if (n < nmax) {
    const float* row = W + (size_t)n * ld + k0;
#pragma unroll
    for (int i = 0; i < 16; ++i) b[i] = (__bf16)row[kmap(i, lane)];
  } else {
#pragma unroll
    for (int i = 0; i < 16; ++i) b[i] = (__bf16)0.0f;
  }
  return b;
}

// ---------------------------------------------------------------------------
// Generic fused GEMM: Y[M,Nout] = epilogue( X[M,K] @ W[Nout,K]^T )
// mode 0: none | 1: bias+elu | 2: bias+bn+elu | 3: bias only
// One wave per 16x64 output strip (4 accumulators): one A-fragment load feeds
// 4 WMMAs per k-step. Fast path (strip fully in range -- every GEMM here
// except the N=1 head) has zero divergence: no exec save/restore, accumulators
// stay pinned in their even-aligned VGPR quads across the whole k-loop.
// M, K multiples of 16/32 by construction.
// ---------------------------------------------------------------------------
__global__ __launch_bounds__(128)
void wmma_linear_kernel(const float* __restrict__ X, const float* __restrict__ W,
                        const float* __restrict__ bias,
                        const float* __restrict__ bng, const float* __restrict__ bnb,
                        const float* __restrict__ bnm, const float* __restrict__ bnv,
                        float* __restrict__ Y, int M, int K, int Nout, int mode) {
  const int ntiles = (Nout + 15) >> 4;          // 16-wide column tiles
  const int nstrip = (ntiles + 3) >> 2;         // 64-wide strips of 4 tiles
  const int mtiles = M >> 4;
  const int tile   = blockIdx.x * (blockDim.x >> 5) + (threadIdx.x >> 5);
  if (tile >= nstrip * mtiles) return;          // wave-uniform
  const int mt = tile / nstrip, qs = tile % nstrip;
  const int lane = threadIdx.x & 31;

  f32x8 c[4];
#pragma unroll
  for (int j = 0; j < 4; ++j)
#pragma unroll
    for (int v = 0; v < 8; ++v) c[j][v] = 0.0f;

  const float* xrow = X + (size_t)(mt * 16 + (lane & 15)) * K;
  const bool full = (qs * 4 + 4) * 16 <= Nout;  // wave-uniform, loop-invariant

  if (full) {
    // -------- fast path: no guards, no divergence --------
    for (int k0 = 0; k0 < K; k0 += 32) {
      if (k0 + 32 < K) __builtin_prefetch(xrow + k0 + 32, 0, 3);  // global_prefetch_b8
      bf16x16 a = load_a_f32(X, mt * 16, k0, K, lane);
#pragma unroll
      for (int j = 0; j < 4; ++j) {
        bf16x16 b = load_b_w(W, (qs * 4 + j) * 16, k0, K, lane);
        c[j] = __builtin_amdgcn_wmma_f32_16x16x32_bf16(false, a, false, b,
                                                       (short)0, c[j], false, false);
      }
    }
  } else {
    // -------- tail path: partial strip (only the N=1 output head) --------
    for (int k0 = 0; k0 < K; k0 += 32) {
      bf16x16 a = load_a_f32(X, mt * 16, k0, K, lane);
#pragma unroll
      for (int j = 0; j < 4; ++j) {
        const int n0 = (qs * 4 + j) * 16;
        if (n0 < Nout) {                        // wave-uniform guard
          bf16x16 b = load_b_w_guard(W, n0, k0, K, Nout, lane);
          c[j] = __builtin_amdgcn_wmma_f32_16x16x32_bf16(false, a, false, b,
                                                         (short)0, c[j], false, false);
        }
      }
    }
  }

  const int mbase = mt * 16 + ((lane >> 4) << 3);
#pragma unroll
  for (int j = 0; j < 4; ++j) {
    const int n = (qs * 4 + j) * 16 + (lane & 15);
    if (n >= Nout) continue;                    // divergence only after all WMMAs
    float bv = bias ? bias[n] : 0.0f;
    float sc = 1.0f, of = 0.0f;
    if (mode == 2) {
      sc = bng[n] * rsqrtf(bnv[n] + 1e-5f);
      of = bnb[n] - bnm[n] * sc;
    }
#pragma unroll
    for (int v = 0; v < 8; ++v) {
      float y = c[j][v] + bv;
      if (mode == 2) y = y * sc + of;
      if (mode == 1 || mode == 2) y = (y > 0.0f) ? y : (expf(y) - 1.0f);
      Y[(size_t)(mbase + v) * Nout + n] = y;
    }
  }
}

// ---------------------------------------------------------------------------
// GATv2 edge phase (3 passes) + helpers
// ---------------------------------------------------------------------------
static __device__ __forceinline__ void edge_pair(int e, const int* __restrict__ eidx,
                                                 int& s, int& d) {
  if (e < N_EDGES) { s = eidx[e]; d = eidx[N_EDGES + e]; }
  else             { s = d = e - N_EDGES; }     // self loop
}

static __device__ float atomicMaxF(float* addr, float val) {
  int* ai = (int*)addr;
  int old = __float_as_int(*addr);
  while (__int_as_float(old) < val) {
    int assumed = old;
    old = atomicCAS(ai, assumed, __float_as_int(val));
    if (old == assumed) break;
  }
  return __int_as_float(old);
}

__global__ void fill_kernel(float* __restrict__ p, float v, long n) {
  long i = blockIdx.x * (long)blockDim.x + threadIdx.x;
  if (i < n) p[i] = v;
}

// pass 1: logits[e,h] = att_h . leaky(xl[dst]+xr[src]) ; segmax via atomic max
__global__ void edge_logits_kernel(const float* __restrict__ xl, const float* __restrict__ xr,
                                   const float* __restrict__ att, const int* __restrict__ eidx,
                                   float* __restrict__ logits, float* __restrict__ segmax,
                                   int H, int Dh, int stride) {
  int i = blockIdx.x * blockDim.x + threadIdx.x;
  if (i >= EE * H) return;
  int e = i / H, h = i - e * H;
  int s, d; edge_pair(e, eidx, s, d);
  const float* pl = xl + (size_t)d * stride + h * Dh;
  const float* pr = xr + (size_t)s * stride + h * Dh;
  const float* pa = att + h * Dh;
  float acc = 0.0f;
#pragma unroll 4
  for (int k = 0; k < Dh; ++k) {
    float g = pl[k] + pr[k];
    g = (g > 0.0f) ? g : 0.2f * g;              // leaky_relu(0.2)
    acc += g * pa[k];
  }
  logits[i] = acc;
  atomicMaxF(&segmax[d * H + h], acc);
}

// pass 2: w = exp(logit - max[dst]); segsum += w  (w stored in-place)
__global__ void edge_expsum_kernel(const int* __restrict__ eidx, float* __restrict__ logits,
                                   const float* __restrict__ segmax, float* __restrict__ segsum,
                                   int H) {
  int i = blockIdx.x * blockDim.x + threadIdx.x;
  if (i >= EE * H) return;
  int e = i / H, h = i - e * H;
  int s, d; edge_pair(e, eidx, s, d);
  float w = expf(logits[i] - segmax[d * H + h]);
  logits[i] = w;
  atomicAdd(&segsum[d * H + h], w);
}

// pass 3: out[dst] += alpha * xr[src]   (global_atomic_add_f32)
__global__ void edge_scatter_kernel(const float* __restrict__ xr, const int* __restrict__ eidx,
                                    const float* __restrict__ logits,
                                    const float* __restrict__ segsum, float* __restrict__ out,
                                    int H, int Dh, int stride) {
  int i = blockIdx.x * blockDim.x + threadIdx.x;
  if (i >= EE * H) return;
  int e = i / H, h = i - e * H;
  int s, d; edge_pair(e, eidx, s, d);
  float alpha = logits[i] / (segsum[d * H + h] + 1e-16f);
  const float* pr = xr + (size_t)s * stride + h * Dh;
  float* pd = out + (size_t)d * stride + h * Dh;
#pragma unroll 4
  for (int k = 0; k < Dh; ++k) atomicAdd(&pd[k], alpha * pr[k]);
}

__global__ void bias_elu_kernel(float* __restrict__ y, const float* __restrict__ b,
                                int stride, long n) {
  long i = blockIdx.x * (long)blockDim.x + threadIdx.x;
  if (i >= n) return;
  float x = y[i] + b[i % stride];
  y[i] = (x > 0.0f) ? x : (expf(x) - 1.0f);
}

__global__ void gather_stop_kernel(const float* __restrict__ h2, const int* __restrict__ stop,
                                   float* __restrict__ fused) {
  int i = blockIdx.x * blockDim.x + threadIdx.x;
  if (i >= BATCH * 128) return;
  int b = i >> 7, c = i & 127;
  fused[(size_t)b * 192 + c] = h2[(size_t)stop[b] * 128 + c];
}

// ---------------------------------------------------------------------------
// GRU: 2 waves/block, 16 batch rows per wave, full T loop in-kernel.
// Weights in LDS as bf16; h/r/z state in per-wave LDS (f32).
// Per step: 12 output tiles x (1 wmma for x-path + 2 wmma for h-path).
// ---------------------------------------------------------------------------
__global__ __launch_bounds__(64)
void gru_kernel(const float* __restrict__ seq, const float* __restrict__ wih,
                const float* __restrict__ whh, const float* __restrict__ bih,
                const float* __restrict__ bhh, const int* __restrict__ lengths,
                float* __restrict__ fused) {
  __shared__ __bf16 s_wih[192 * 32];            // K padded 16 -> 32 with zeros
  __shared__ __bf16 s_whh[192 * 64];
  __shared__ float  s_bih[192], s_bhh[192];
  __shared__ float  s_h[2][16 * 64], s_r[2][16 * 64], s_z[2][16 * 64];

  const int tid = threadIdx.x;
  for (int i = tid; i < 192 * 32; i += 64) {
    int r = i >> 5, k = i & 31;
    s_wih[i] = (k < FSEQ) ? (__bf16)wih[r * FSEQ + k] : (__bf16)0.0f;
  }
  for (int i = tid; i < 192 * 64; i += 64) s_whh[i] = (__bf16)whh[i];
  for (int i = tid; i < 192; i += 64) { s_bih[i] = bih[i]; s_bhh[i] = bhh[i]; }
  __syncthreads();

  const int w = tid >> 5, lane = tid & 31;
  const int b0 = blockIdx.x * 32 + w * 16;
  float* hbuf = s_h[w]; float* rbuf = s_r[w]; float* zbuf = s_z[w];
  for (int i = lane; i < 1024; i += 32) hbuf[i] = 0.0f;

  const int mrow = (lane >> 4) << 3;            // D-row base (0 or 8)
  const int nloc = lane & 15;
  int lens[8];
#pragma unroll
  for (int v = 0; v < 8; ++v) lens[v] = lengths[b0 + mrow + v];

  for (int t = 0; t < TMAX; ++t) {
    bf16x16 ax, ah0, ah1;
    {
      const int m = lane & 15;
      const float* srow = seq + ((size_t)(b0 + m) * TMAX + t) * FSEQ;
      const int koff = (lane >> 4) << 3;
#pragma unroll
      for (int i = 0; i < 16; ++i) ax[i] = (i < 8) ? (__bf16)srow[i + koff] : (__bf16)0.0f;
#pragma unroll
      for (int i = 0; i < 16; ++i) ah0[i] = (__bf16)hbuf[m * 64 + kmap(i, lane)];
#pragma unroll
      for (int i = 0; i < 16; ++i) ah1[i] = (__bf16)hbuf[m * 64 + 32 + kmap(i, lane)];
    }
#pragma unroll
    for (int tile = 0; tile < 12; ++tile) {
      const int g = tile * 16 + nloc;           // gate index 0..191
      bf16x16 bx, bh0, bh1;
#pragma unroll
      for (int i = 0; i < 16; ++i) bx[i]  = s_wih[g * 32 + kmap(i, lane)];
#pragma unroll
      for (int i = 0; i < 16; ++i) bh0[i] = s_whh[g * 64 + kmap(i, lane)];
#pragma unroll
      for (int i = 0; i < 16; ++i) bh1[i] = s_whh[g * 64 + 32 + kmap(i, lane)];
      f32x8 cx, ch;
#pragma unroll
      for (int v = 0; v < 8; ++v) { cx[v] = 0.0f; ch[v] = 0.0f; }
      cx = __builtin_amdgcn_wmma_f32_16x16x32_bf16(false, ax,  false, bx,  (short)0, cx, false, false);
      ch = __builtin_amdgcn_wmma_f32_16x16x32_bf16(false, ah0, false, bh0, (short)0, ch, false, false);
      ch = __builtin_amdgcn_wmma_f32_16x16x32_bf16(false, ah1, false, bh1, (short)0, ch, false, false);
#pragma unroll
      for (int v = 0; v < 8; ++v) {
        const int m = mrow + v;
        const float xg = cx[v] + s_bih[g];
        const float hg = ch[v] + s_bhh[g];
        if (tile < 4) {
          rbuf[m * 64 + g] = 1.0f / (1.0f + expf(-(xg + hg)));
        } else if (tile < 8) {
          zbuf[m * 64 + (g - 64)] = 1.0f / (1.0f + expf(-(xg + hg)));
        } else {
          const int col = g - 128;
          const float r = rbuf[m * 64 + col];
          const float z = zbuf[m * 64 + col];
          const float hold = hbuf[m * 64 + col];
          const float nn = tanhf(xg + r * hg);
          const float hnew = (1.0f - z) * nn + z * hold;
          hbuf[m * 64 + col] = (t < lens[v]) ? hnew : hold;
        }
      }
    }
  }
  for (int i = lane; i < 1024; i += 32) {
    const int m = i >> 6, col = i & 63;
    fused[(size_t)(b0 + m) * 192 + 128 + col] = hbuf[i];
  }
}

// ---------------------------------------------------------------------------
// Host orchestration
// ---------------------------------------------------------------------------
extern "C" void kernel_launch(void* const* d_in, const int* in_sizes, int n_in,
                              void* d_out, int out_size, void* d_ws, size_t ws_size,
                              hipStream_t stream) {
  const float* x    = (const float*)d_in[0];
  const float* seq  = (const float*)d_in[1];
  const float* g1wl = (const float*)d_in[2];
  const float* g1wr = (const float*)d_in[3];
  const float* g1at = (const float*)d_in[4];
  const float* g1b  = (const float*)d_in[5];
  const float* g2wl = (const float*)d_in[6];
  const float* g2wr = (const float*)d_in[7];
  const float* g2at = (const float*)d_in[8];
  const float* g2b  = (const float*)d_in[9];
  const float* wih  = (const float*)d_in[10];
  const float* whh  = (const float*)d_in[11];
  const float* bih  = (const float*)d_in[12];
  const float* bhh  = (const float*)d_in[13];
  const float* fc1w = (const float*)d_in[14];
  const float* fc1b = (const float*)d_in[15];
  const float* bn1g = (const float*)d_in[16];
  const float* bn1b = (const float*)d_in[17];
  const float* bn1m = (const float*)d_in[18];
  const float* bn1v = (const float*)d_in[19];
  const float* fc2w = (const float*)d_in[20];
  const float* fc2b = (const float*)d_in[21];
  const float* bn2g = (const float*)d_in[22];
  const float* bn2b = (const float*)d_in[23];
  const float* bn2m = (const float*)d_in[24];
  const float* bn2v = (const float*)d_in[25];
  const float* fc3w = (const float*)d_in[26];
  const float* fc3b = (const float*)d_in[27];
  const float* fcow = (const float*)d_in[28];
  const float* fcob = (const float*)d_in[29];
  const int* eidx = (const int*)d_in[30];
  const int* lens = (const int*)d_in[31];
  const int* stop = (const int*)d_in[32];

  char* ws = (char*)d_ws;
  const size_t S1 = (size_t)N_NODES * 256 * 4;              // 51.2 MB
  float* xl1   = (float*)(ws + 0);
  float* xr1   = (float*)(ws + S1);
  float* acc1  = (float*)(ws + 2 * S1);                     // -> h1
  float* log1  = (float*)(ws + 3 * S1);                     // EE*4 f32
  float* smax1 = (float*)(ws + 3 * S1 + (size_t)EE * 4 * 4);
  float* ssum1 = (float*)((char*)smax1 + (size_t)N_NODES * 4 * 4);
  // layer 2 reuses layer-1 regions (xl1/xr1 dead once h1 exists)
  float* xl2   = (float*)(ws + 0);
  float* xr2   = (float*)(ws + (size_t)N_NODES * 128 * 4);
  float* acc2  = (float*)(ws + S1);                         // -> h2
  float* log2  = (float*)(ws + S1 + (size_t)N_NODES * 128 * 4);
  float* smax2 = (float*)((char*)log2 + (size_t)EE * 4);
  float* ssum2 = (float*)((char*)smax2 + (size_t)N_NODES * 4);
  // fusion stage reuses xl2/xr2 region (dead after edge pass 3)
  float* fused = (float*)(ws + 0);                          // [4096,192]
  float* a1    = (float*)(ws + 4  * 1000000);               // [4096,256]
  float* a2    = (float*)(ws + 9  * 1000000);               // [4096,128]
  float* a3    = (float*)(ws + 12 * 1000000);               // [4096,64]

  auto gemm = [&](const float* X, const float* W, const float* b,
                  const float* bg, const float* bb, const float* bm, const float* bv,
                  float* Y, int M, int K, int Nout, int mode) {
    int ntiles = (Nout + 15) / 16;
    int nstrip = (ntiles + 3) / 4;
    int tiles  = (M / 16) * nstrip;
    wmma_linear_kernel<<<(tiles + 3) / 4, 128, 0, stream>>>(X, W, b, bg, bb, bm, bv,
                                                            Y, M, K, Nout, mode);
  };
  auto fill = [&](float* p, float v, long n) {
    fill_kernel<<<(int)((n + 255) / 256), 256, 0, stream>>>(p, v, n);
  };

  // ---- GAT layer 1 (4 heads x 64) ----
  gemm(x, g1wl, nullptr, nullptr, nullptr, nullptr, nullptr, xl1, N_NODES, 128, 256, 0);
  gemm(x, g1wr, nullptr, nullptr, nullptr, nullptr, nullptr, xr1, N_NODES, 128, 256, 0);
  fill(smax1, -3.0e38f, (long)N_NODES * 4);
  fill(ssum1, 0.0f, (long)N_NODES * 4);
  fill(acc1, 0.0f, (long)N_NODES * 256);
  {
    int th = EE * 4, bl = (th + 255) / 256;
    edge_logits_kernel<<<bl, 256, 0, stream>>>(xl1, xr1, g1at, eidx, log1, smax1, 4, 64, 256);
    edge_expsum_kernel<<<bl, 256, 0, stream>>>(eidx, log1, smax1, ssum1, 4);
    edge_scatter_kernel<<<bl, 256, 0, stream>>>(xr1, eidx, log1, ssum1, acc1, 4, 64, 256);
  }
  bias_elu_kernel<<<(int)(((long)N_NODES * 256 + 255) / 256), 256, 0, stream>>>(
      acc1, g1b, 256, (long)N_NODES * 256);

  // ---- GAT layer 2 (1 head x 128) ----
  gemm(acc1, g2wl, nullptr, nullptr, nullptr, nullptr, nullptr, xl2, N_NODES, 256, 128, 0);
  gemm(acc1, g2wr, nullptr, nullptr, nullptr, nullptr, nullptr, xr2, N_NODES, 256, 128, 0);
  fill(smax2, -3.0e38f, N_NODES);
  fill(ssum2, 0.0f, N_NODES);
  fill(acc2, 0.0f, (long)N_NODES * 128);
  {
    int th = EE, bl = (th + 255) / 256;
    edge_logits_kernel<<<bl, 256, 0, stream>>>(xl2, xr2, g2at, eidx, log2, smax2, 1, 128, 128);
    edge_expsum_kernel<<<bl, 256, 0, stream>>>(eidx, log2, smax2, ssum2, 1);
    edge_scatter_kernel<<<bl, 256, 0, stream>>>(xr2, eidx, log2, ssum2, acc2, 1, 128, 128);
  }
  bias_elu_kernel<<<(int)(((long)N_NODES * 128 + 255) / 256), 256, 0, stream>>>(
      acc2, g2b, 128, (long)N_NODES * 128);

  // ---- Fusion: stop embedding gather + GRU last hidden ----
  gather_stop_kernel<<<(BATCH * 128 + 255) / 256, 256, 0, stream>>>(acc2, stop, fused);
  gru_kernel<<<BATCH / 32, 64, 0, stream>>>(seq, wih, whh, bih, bhh, lens, fused);

  // ---- MLP head ----
  gemm(fused, fc1w, fc1b, bn1g, bn1b, bn1m, bn1v, a1, BATCH, 192, 256, 2);
  gemm(a1, fc2w, fc2b, bn2g, bn2b, bn2m, bn2v, a2, BATCH, 256, 128, 2);
  gemm(a2, fc3w, fc3b, nullptr, nullptr, nullptr, nullptr, a3, BATCH, 128, 64, 1);
  gemm(a3, fcow, fcob, nullptr, nullptr, nullptr, nullptr, (float*)d_out, BATCH, 64, 1, 3);
}
